// Correlation_21715354649744
// MI455X (gfx1250) — matrix operand order repaired and checked by reading, not compiled, not run on hardware
//
#include <hip/hip_runtime.h>
#include <hip/hip_bf16.h>
#include <stdint.h>

typedef __attribute__((ext_vector_type(16))) _Float16 v16h;
typedef __attribute__((ext_vector_type(8)))  _Float16 v8h;
typedef __attribute__((ext_vector_type(8)))  float    v8f;
typedef __attribute__((ext_vector_type(4)))  unsigned int u32x4;
typedef __attribute__((ext_vector_type(8)))  int      i32x8;
typedef __attribute__((ext_vector_type(4)))  int      i32x4;

#define CB 16
#define CC 256
#define CH 48
#define CW 64
#define NI 21                 // displacements per axis
#define RS 264                // LDS row stride in halves (256 + 8 pad -> 528B, conflict-free)
#define AROWS 64
#define BROWS 136             // packed B columns: w' in [-24, 112)
#define BOFF  (AROWS * RS)    // halves
#define HALF_PER_ROW (CW * CC) // 16384 halves = 32KB per (b,h) feature row-block

// ---------------------------------------------------------------------------
// TDM: one tensor_load_to_lds copies a whole 32KB [w][c] f16 row-block into
// LDS, inserting 16B of padding after every 512B row (pad_interval=128 dwords,
// pad_amount=4 dwords). Descriptor built per CDNA5 ISA.
// This toolchain exposes the 6-arg builtin: (g0, g1, g2, g3, g_extra, cpol).
// ---------------------------------------------------------------------------
__device__ __forceinline__ void tdm_load_row(const _Float16* gsrc, uint32_t lds_byte_off) {
  uint64_t ga = (uint64_t)(uintptr_t)gsrc;
  u32x4 g0;
  g0[0] = 1u;                                                  // count=1, user descriptor
  g0[1] = lds_byte_off;                                        // lds_addr (bytes)
  g0[2] = (uint32_t)ga;                                        // global_addr lo
  g0[3] = (uint32_t)((ga >> 32) & 0x01FFFFFFu) | (2u << 30);   // global_addr hi | type=2
  i32x8 g1;
  g1[0] = (1 << 16)        // data_size = 1 (2-byte elements)
        | (1 << 20)        // pad_enable
        | (6 << 22)        // pad_interval: 128 dwords (512B) between pads
        | (3 << 25);       // pad_amount: 4 dwords (16B)
  g1[1] = (int)(((unsigned)HALF_PER_ROW & 0xFFFFu) << 16);     // tensor_dim0 [15:0]
  g1[2] = (int)(((unsigned)HALF_PER_ROW >> 16) & 0xFFFFu)      // tensor_dim0 [31:16]
        | (1 << 16);                                           // tensor_dim1 = 1
  g1[3] = (int)(((unsigned)HALF_PER_ROW & 0xFFFFu) << 16);     // tile_dim0 = 16384
  g1[4] = 1;                                                   // tile_dim1 = 1, tile_dim2 = 0
  g1[5] = HALF_PER_ROW;                                        // tensor_dim0_stride lo32
  g1[6] = 0;
  g1[7] = 0;
  i32x4 z4 = {0, 0, 0, 0};
  i32x8 z8 = {0, 0, 0, 0, 0, 0, 0, 0};
  __builtin_amdgcn_tensor_load_to_lds(g0, g1, z4, z4, z8, 0);
}

// ---------------------------------------------------------------------------
// Pre-pass: [B,C,H,W] f32  ->  [B,H,W,C] f16 (c-innermost) for both inputs.
// Coalesced f32 reads over w, LDS transpose, coalesced 16B f16 stores.
// ---------------------------------------------------------------------------
__global__ void __launch_bounds__(256) cvt_to_f16_whc(const float* __restrict__ in1,
                                                      const float* __restrict__ in2,
                                                      _Float16* __restrict__ ws,
                                                      size_t tensor_halves) {
  __shared__ __align__(16) _Float16 tile[CW * RS];
  const int id  = blockIdx.x;
  const int src = id & 1;
  const int bh  = id >> 1;
  const int b = bh / CH;
  const int h = bh - b * CH;
  const float* sp = (src ? in2 : in1) + (size_t)b * (CC * CH * CW) + (size_t)h * CW;
  _Float16* dst = ws + (size_t)src * tensor_halves + (size_t)bh * HALF_PER_ROW;

  for (int j = threadIdx.x; j < CC * CW; j += 256) {
    const int c = j >> 6, w = j & 63;                          // lanes -> consecutive w
    tile[w * RS + c] = (_Float16)sp[(size_t)c * (CH * CW) + w];
  }
  __syncthreads();
  for (int j = threadIdx.x; j < (CW * CC) / 8; j += 256) {
    const int w = j >> 5, cg = j & 31;
    *(v8h*)(dst + w * CC + cg * 8) = *(const v8h*)(tile + w * RS + cg * 8);
  }
}

// ---------------------------------------------------------------------------
// Main kernel: one workgroup per (b,h); 12 waves = 4 M-tiles x 3 N-tiles.
// For each dy: banded Gram matrix via v_wmma_f32_16x16x32_f16, K=C=256.
// ---------------------------------------------------------------------------
__global__ void __launch_bounds__(384, 1) corr_wmma(const _Float16* __restrict__ w1,
                                                    const _Float16* __restrict__ w2,
                                                    float* __restrict__ out) {
  __shared__ __align__(16) _Float16 lds[(AROWS + BROWS) * RS];   // 105,600 B
  const int tid  = threadIdx.x;
  const int lane = tid & 31;
  const int wv   = tid >> 5;        // wave 0..11
  const int mt   = wv & 3;          // M-tile: parity p, half
  const int t    = wv >> 2;         // N-tile 0..2
  const int p    = mt & 1;
  const int half = mt >> 1;
  const int bh = blockIdx.x;
  const int b  = bh / CH;
  const int h  = bh - b * CH;

  // Zero entire B region once: out-of-image columns stay zero forever;
  // in-image columns (rows 0..63) are rewritten by the TDM each iteration.
  {
    v8h z = {};
    for (int j = tid; j < (BROWS * RS) / 8; j += 384)
      *(v8h*)(lds + BOFF + j * 8) = z;
  }

  // Kick off A-row DMA (covered by the first s_wait_tensorcnt below).
  if (wv == 0)
    tdm_load_row(w1 + (size_t)bh * HALF_PER_ROW, (uint32_t)(uintptr_t)(&lds[0]));

  // Per-lane WMMA fragment addresses (ISA 16-bit A/B layouts, wave32).
  const int nm   = lane & 15;
  const int hiL  = lane >> 4;
  const int w_row = 32 * half + 2 * nm + p;                 // A: row = w, chunks K +0/+16
  const _Float16* Ap = lds + w_row * RS + hiL * 8;
  const int cb   = 32 * half + p - 20 + 32 * t;             // first B column of this N-tile
  const int brow = cb + 24 + 2 * nm;                        // packed B row (w' + 24)
  const _Float16* Bp = lds + BOFF + brow * RS + hiL * 8;    // B: chunks K +0/+16, same as A
  const int sx_base = 32 * t - 20 + 2 * nm;                 // sx = sx_base - 2*m

  for (int iy = 0; iy < NI; ++iy) {
    const int h2 = h + 2 * iy - 20;
    const bool valid = (h2 >= 0) && (h2 < CH);

    __syncthreads();                 // all waves done reading previous B
    if (valid && wv == 0) {
      tdm_load_row(w2 + ((size_t)b * CH + h2) * HALF_PER_ROW,
                   (uint32_t)(uintptr_t)(&lds[BOFF]));
      __builtin_amdgcn_s_wait_tensorcnt(0);
    }
    __syncthreads();                 // B (and, first time, A) resident

    if (!valid) {                    // whole dy row is zero -> store zeros
      for (int j = tid; j < NI * CW; j += 384) {
        const int ix = j >> 6, w = j & 63;
        out[(((size_t)b * (NI * NI) + iy * NI + ix) * CH + h) * CW + w] = 0.0f;
      }
      continue;
    }

    v8f acc = {};
#pragma unroll
    for (int ks = 0; ks < 8; ++ks) {                         // K = 256 = 8 x 32
      v8h alo = *(const v8h*)(Ap + ks * 32);
      v8h ahi = *(const v8h*)(Ap + ks * 32 + 16);
      v8h blo = *(const v8h*)(Bp + ks * 32);
      v8h bhi = *(const v8h*)(Bp + ks * 32 + 16);
      v16h a  = __builtin_shufflevector(alo, ahi, 0,1,2,3,4,5,6,7,8,9,10,11,12,13,14,15);
      v16h bm = __builtin_shufflevector(blo, bhi, 0,1,2,3,4,5,6,7,8,9,10,11,12,13,14,15);
      acc = __builtin_amdgcn_wmma_f32_16x16x32_f16(false, a, false, bm,
                                                   (short)0, acc, false, false);
    }

    // D layout: N = lane&15, M = r + 8*(lane>=16). Store shifts in [-20,20].
#pragma unroll
    for (int r = 0; r < 8; ++r) {
      const int m  = r + 8 * hiL;
      const int sx = sx_base - 2 * m;
      if ((unsigned)(sx + 20) <= 40u) {
        const int ix = (sx + 20) >> 1;
        const int w  = 32 * half + 2 * m + p;
        out[(((size_t)b * (NI * NI) + iy * NI + ix) * CH + h) * CW + w] =
            acc[r] * (1.0f / 256.0f);
      }
    }
  }
}

extern "C" void kernel_launch(void* const* d_in, const int* in_sizes, int n_in,
                              void* d_out, int out_size, void* d_ws, size_t ws_size,
                              hipStream_t stream) {
  const float* in1 = (const float*)d_in[0];
  const float* in2 = (const float*)d_in[1];
  _Float16* ws = (_Float16*)d_ws;
  const size_t tensor_halves = (size_t)CB * CH * CW * CC;   // 12.6M halves each
  cvt_to_f16_whc<<<dim3(CB * CH * 2), dim3(256), 0, stream>>>(in1, in2, ws, tensor_halves);
  corr_wmma<<<dim3(CB * CH), dim3(384), 0, stream>>>(ws, ws + tensor_halves, (float*)d_out);
}
